// StableAttention_20143396618787
// MI455X (gfx1250) — matrix operand order repaired and checked by reading, not compiled, not run on hardware
//
#include <hip/hip_runtime.h>

#define T_SEQ  4096
#define DMODEL 1024
#define NHEAD  16
#define DKHEAD 64
#define SCALE_QK 0.125f      // 1/sqrt(64)
#define NEGBIG  -1e9f

typedef __attribute__((ext_vector_type(16))) __bf16         v16bf;
typedef __attribute__((ext_vector_type(8)))  float          v8f;
typedef __attribute__((ext_vector_type(8)))  unsigned short u16x8;
typedef __attribute__((ext_vector_type(4)))  unsigned short u16x4;

union FragU { v16bf v; u16x8 h[2]; };

__device__ __forceinline__ v8f v8f_zero() {
  v8f z = {0.f,0.f,0.f,0.f,0.f,0.f,0.f,0.f};
  return z;
}

__device__ __forceinline__ unsigned short f32_bf16(float f) {
  unsigned u = __float_as_uint(f);
  u += 0x7fffu + ((u >> 16) & 1u);          // round-to-nearest-even
  return (unsigned short)(u >> 16);
}

// ---- CDNA5 async copy: 16B global -> LDS per lane, tracked by ASYNCcnt -----------------
__device__ __forceinline__ void async_cp16(unsigned lds_byte_off, const unsigned short* gsrc) {
  unsigned long long ga = (unsigned long long)(const void*)gsrc;  // global aperture: identity
  asm volatile("global_load_async_to_lds_b128 %0, %1, off"
               :: "v"(lds_byte_off), "v"(ga) : "memory");
}
__device__ __forceinline__ void async_wait0() {
  asm volatile("s_wait_asynccnt 0x0" ::: "memory");
}
__device__ __forceinline__ unsigned lds_off(const unsigned short* p) {
  // generic LDS pointer: addr[31:0] is the wave-relative LDS byte offset (ISA 10.2)
  return (unsigned)(unsigned long long)(const void*)p;
}

// A-fragment (16x32 bf16, MxK): lane<16 holds row M=lane, K={0..7,16..23};
// lane>=16 holds row M=lane-16, K={8..15,24..31}. Two 16B loads.
__device__ __forceinline__ v16bf load_a_frag(const unsigned short* base, int lane, int ldt) {
  int r = lane & 15, h = lane >> 4;
  FragU f;
  f.h[0] = *(const u16x8*)(base + (size_t)r * ldt + h * 8);
  f.h[1] = *(const u16x8*)(base + (size_t)r * ldt + 16 + h * 8);
  return f.v;
}

// B-fragment for C = A * B^T where the "B^T source" is row-major [N][K]:
// lane<16 holds source-row n=lane, K={0..7,16..23}; lane>=16 holds K={8..15,24..31}. Two 16B loads.
__device__ __forceinline__ v16bf load_bt_frag(const unsigned short* base, int lane, size_t ldt) {
  int r = lane & 15, h = lane >> 4;
  FragU f;
  f.h[0] = *(const u16x8*)(base + (size_t)r * ldt + h * 8);
  f.h[1] = *(const u16x8*)(base + (size_t)r * ldt + 16 + h * 8);
  return f.v;
}

#define LDT 40   // 32 bf16 + 8 pad per LDS tile row (80B rows -> conflict-free b128 reads)

// Stage a 128x32 fp32 tile (global row stride 1024 floats) into LDS as bf16.
__device__ __forceinline__ void stage_tile_f32(const float* __restrict__ g,
                                               unsigned short* lds, int tid) {
  int tr = tid >> 3;
  int tc = (tid & 7) << 2;
  #pragma unroll
  for (int i = 0; i < 4; ++i) {
    int r = tr + i * 32;
    float4 f = *(const float4*)(g + (size_t)r * 1024 + tc);
    u16x4 o = { f32_bf16(f.x), f32_bf16(f.y), f32_bf16(f.z), f32_bf16(f.w) };
    *(u16x4*)(lds + r * LDT + tc) = o;
  }
}

// 128x128 C-tile, K=1024, A [Mx1024] fp32, Bt [Nx1024] fp32 (computes A*Bt^T).
// 8 waves: wave (wm 0..3, wn 0..1) owns 32x64 = 2x4 wmma accumulators.
__device__ __forceinline__ void gemm_core(const float* __restrict__ Abase,
                                          const float* __restrict__ Bbase,
                                          unsigned short* ldsA, unsigned short* ldsB,
                                          v8f acc[2][4]) {
  int tid  = threadIdx.x;
  int lane = tid & 31, wid = tid >> 5;
  int wm = wid >> 1, wn = wid & 1;
  for (int kt = 0; kt < 1024; kt += 32) {
    stage_tile_f32(Abase + kt, ldsA, tid);
    stage_tile_f32(Bbase + kt, ldsB, tid);
    if (kt + 32 < 1024) {   // prefetch next k-tile into cache (global_prefetch_b8)
      size_t po = (size_t)(tid >> 3) * 1024 + ((tid & 7) << 2);
      __builtin_prefetch(Abase + kt + 32 + po, 0, 1);
      __builtin_prefetch(Bbase + kt + 32 + po, 0, 1);
    }
    __syncthreads();
    v16bf af[2], bf4[4];
    #pragma unroll
    for (int mi = 0; mi < 2; ++mi)
      af[mi] = load_a_frag(ldsA + (wm * 32 + mi * 16) * LDT, lane, LDT);
    #pragma unroll
    for (int ni = 0; ni < 4; ++ni)
      bf4[ni] = load_bt_frag(ldsB + (wn * 64 + ni * 16) * LDT, lane, LDT);
    #pragma unroll
    for (int mi = 0; mi < 2; ++mi)
      #pragma unroll
      for (int ni = 0; ni < 4; ++ni)
        acc[mi][ni] = __builtin_amdgcn_wmma_f32_16x16x32_bf16(
            false, af[mi], false, bf4[ni], (short)0, acc[mi][ni], false, false);
    __syncthreads();
  }
}

// ---------------- Kernel 1: QKV = x @ Wqkv^T, scattered into Q/K/V^T (bf16) -------------
__global__ __launch_bounds__(256)
void qkv_gemm_kernel(const float* __restrict__ x, const float* __restrict__ Wqkv,
                     unsigned short* __restrict__ Q, unsigned short* __restrict__ Kb,
                     unsigned short* __restrict__ Vt) {
  __shared__ unsigned short ldsA[128 * LDT];
  __shared__ unsigned short ldsB[128 * LDT];
  int row0 = blockIdx.x * 128;
  int col0 = blockIdx.y * 128;
  v8f acc[2][4];
  #pragma unroll
  for (int mi = 0; mi < 2; ++mi)
    #pragma unroll
    for (int ni = 0; ni < 4; ++ni) acc[mi][ni] = v8f_zero();

  gemm_core(x + (size_t)row0 * 1024, Wqkv + (size_t)col0 * 1024, ldsA, ldsB, acc);

  int lane = threadIdx.x & 31, wid = threadIdx.x >> 5;
  int wm = wid >> 1, wn = wid & 1;
  int half = lane >> 4, nn = lane & 15;
  #pragma unroll
  for (int mi = 0; mi < 2; ++mi)
    #pragma unroll
    for (int ni = 0; ni < 4; ++ni)
      #pragma unroll
      for (int v = 0; v < 8; ++v) {
        int t = row0 + wm * 32 + mi * 16 + half * 8 + v;
        int e = col0 + wn * 64 + ni * 16 + nn;
        unsigned short bv = f32_bf16(acc[mi][ni][v]);
        int which = e >> 10;                 // 0=Q 1=K 2=V
        int hh    = (e >> 6) & (NHEAD - 1);
        int dk    = e & (DKHEAD - 1);
        if (which == 0)       Q [((size_t)(hh * T_SEQ + t) << 6) + dk] = bv;
        else if (which == 1)  Kb[((size_t)(hh * T_SEQ + t) << 6) + dk] = bv;
        else                  Vt[(size_t)(hh * DKHEAD + dk) * T_SEQ + t] = bv;
      }
}

// ---------------- Kernel 2: causal flash attention -------------------------------------
// Block = 8 waves = 128 query rows. K/V^T tiles are staged cooperatively into LDS with
// double buffering via async global->LDS copies; each wave owns 16 query rows.
#define LDK 72   // 64 + 8 pad (bf16 elems) for K tile rows
#define LDV 40   // 32 + 8 pad for V^T tile rows

__device__ __forceinline__ void stage_kv(const unsigned short* __restrict__ Kh,
                                         const unsigned short* __restrict__ Vh,
                                         unsigned short* kdst, unsigned short* vdst,
                                         int s0, int tid) {
  {  // K tile: 32 rows x 64 cols bf16 = 256 x 16B chunks
    int row = tid >> 3, c = (tid & 7) << 3;
    async_cp16(lds_off(kdst + row * LDK + c), Kh + ((size_t)(s0 + row) << 6) + c);
  }
  {  // V^T tile: 64 rows x 32 cols bf16 = 256 x 16B chunks
    int row = tid >> 2, c = (tid & 3) << 3;
    async_cp16(lds_off(vdst + row * LDV + c), Vh + (size_t)row * T_SEQ + s0 + c);
  }
}

__global__ __launch_bounds__(256)
void attn_kernel(const unsigned short* __restrict__ Q, const unsigned short* __restrict__ Kb,
                 const unsigned short* __restrict__ Vt, float* __restrict__ O) {
  __shared__ unsigned short ldsK[2][32 * LDK];
  __shared__ unsigned short ldsV[2][64 * LDV];
  __shared__ unsigned short pbuf[8][16 * LDT];   // per-wave P re-layout scratch
  int h     = blockIdx.y;
  int qbase = blockIdx.x * 128;
  int tid   = threadIdx.x;
  int lane  = tid & 31, wid = tid >> 5;
  int m0    = qbase + wid * 16;
  int hf    = lane >> 4, nn = lane & 15;
  const unsigned short* Qh = Q  + ((size_t)h * T_SEQ << 6);
  const unsigned short* Kh = Kb + ((size_t)h * T_SEQ << 6);
  const unsigned short* Vh = Vt + ((size_t)h << 6) * T_SEQ;
  unsigned short* pw = pbuf[wid];

  v16bf qf[2];
  #pragma unroll
  for (int kk = 0; kk < 2; ++kk)
    qf[kk] = load_a_frag(Qh + ((size_t)m0 << 6) + kk * 32, lane, 64);

  v8f oacc[4];
  #pragma unroll
  for (int d = 0; d < 4; ++d) oacc[d] = v8f_zero();
  float rmax[8], rsum[8];
  #pragma unroll
  for (int v = 0; v < 8; ++v) { rmax[v] = -__builtin_inff(); rsum[v] = 0.f; }

  int nsteps = (qbase + 128) >> 5;              // block's causal extent in 32-wide steps
  stage_kv(Kh, Vh, &ldsK[0][0], &ldsV[0][0], 0, tid);
  async_wait0();
  __syncthreads();

  for (int it = 0; it < nsteps; ++it) {
    int cur = it & 1;
    if (it + 1 < nsteps)                        // prefetch next tiles into other buffer
      stage_kv(Kh, Vh, &ldsK[cur ^ 1][0], &ldsV[cur ^ 1][0], (it + 1) << 5, tid);
    int s0 = it << 5;
    if (s0 < m0 + 16) {                         // wave-uniform causal guard
      // S(16x32) = Q(16x64) * K^T
      v8f sacc[2];
      sacc[0] = v8f_zero(); sacc[1] = v8f_zero();
      #pragma unroll
      for (int ni = 0; ni < 2; ++ni)
        #pragma unroll
        for (int kk = 0; kk < 2; ++kk) {
          v16bf kf = load_bt_frag(&ldsK[cur][(ni * 16) * LDK + kk * 32], lane, LDK);
          sacc[ni] = __builtin_amdgcn_wmma_f32_16x16x32_bf16(
              false, qf[kk], false, kf, (short)0, sacc[ni], false, false);
        }
      // online softmax: row M = hf*8 + v lives across the 16 lanes of this half-wave
      float fac[8];
      #pragma unroll
      for (int v = 0; v < 8; ++v) {
        int t = m0 + hf * 8 + v;
        float e0 = fminf(fmaxf(sacc[0][v] * SCALE_QK, -10.f), 10.f);
        float e1 = fminf(fmaxf(sacc[1][v] * SCALE_QK, -10.f), 10.f);
        if (s0 + nn > t)      e0 += NEGBIG;     // strict causal mask
        if (s0 + 16 + nn > t) e1 += NEGBIG;
        float mx = fmaxf(e0, e1);
        #pragma unroll
        for (int m = 1; m < 16; m <<= 1) mx = fmaxf(mx, __shfl_xor(mx, m, 32));
        mx = fmaxf(rmax[v], mx);
        fac[v] = __expf(rmax[v] - mx);
        rmax[v] = mx;
        float p0 = __expf(e0 - mx), p1 = __expf(e1 - mx);
        float ps = p0 + p1;
        #pragma unroll
        for (int m = 1; m < 16; m <<= 1) ps += __shfl_xor(ps, m, 32);
        rsum[v] = rsum[v] * fac[v] + ps;
        pw[(hf * 8 + v) * LDT + nn]      = f32_bf16(p0);   // P row-major in LDS
        pw[(hf * 8 + v) * LDT + 16 + nn] = f32_bf16(p1);
      }
      #pragma unroll
      for (int d = 0; d < 4; ++d)
        #pragma unroll
        for (int v = 0; v < 8; ++v) oacc[d][v] *= fac[v];
      // O(16x64) += P(16x32) * (V^T)^T  -- V^T rows are contiguous in s
      v16bf pf = load_a_frag(pw, lane, LDT);
      #pragma unroll
      for (int d = 0; d < 4; ++d) {
        v16bf vf = load_bt_frag(&ldsV[cur][(d * 16) * LDV], lane, LDV);
        oacc[d] = __builtin_amdgcn_wmma_f32_16x16x32_bf16(
            false, pf, false, vf, (short)0, oacc[d], false, false);
      }
    }
    async_wait0();                              // staged tiles resident before swap
    __syncthreads();
  }
  #pragma unroll
  for (int v = 0; v < 8; ++v) {
    float inv = 1.0f / rsum[v];
    int t = m0 + hf * 8 + v;
    #pragma unroll
    for (int d = 0; d < 4; ++d)
      O[(size_t)t * DMODEL + h * DKHEAD + d * 16 + nn] = oacc[d][v] * inv;
  }
}

// ---------------- Kernel 3: out = O @ Wout^T (fp32 result) ------------------------------
__global__ __launch_bounds__(256)
void out_gemm_kernel(const float* __restrict__ O, const float* __restrict__ Wout,
                     float* __restrict__ out) {
  __shared__ unsigned short ldsA[128 * LDT];
  __shared__ unsigned short ldsB[128 * LDT];
  int row0 = blockIdx.x * 128;
  int col0 = blockIdx.y * 128;
  v8f acc[2][4];
  #pragma unroll
  for (int mi = 0; mi < 2; ++mi)
    #pragma unroll
    for (int ni = 0; ni < 4; ++ni) acc[mi][ni] = v8f_zero();

  gemm_core(O + (size_t)row0 * 1024, Wout + (size_t)col0 * 1024, ldsA, ldsB, acc);

  int lane = threadIdx.x & 31, wid = threadIdx.x >> 5;
  int wm = wid >> 1, wn = wid & 1;
  int half = lane >> 4, nn = lane & 15;
  #pragma unroll
  for (int mi = 0; mi < 2; ++mi)
    #pragma unroll
    for (int ni = 0; ni < 4; ++ni)
      #pragma unroll
      for (int v = 0; v < 8; ++v) {
        int t = row0 + wm * 32 + mi * 16 + half * 8 + v;
        int e = col0 + wn * 64 + ni * 16 + nn;
        out[(size_t)t * DMODEL + e] = acc[mi][ni][v];
      }
}

extern "C" void kernel_launch(void* const* d_in, const int* in_sizes, int n_in,
                              void* d_out, int out_size, void* d_ws, size_t ws_size,
                              hipStream_t stream) {
  (void)in_sizes; (void)n_in; (void)out_size; (void)ws_size;
  const float* x    = (const float*)d_in[0];   // [1,4096,1024]
  const float* Wqkv = (const float*)d_in[1];   // [3072,1024]
  const float* Wout = (const float*)d_in[2];   // [1024,1024]
  float* out = (float*)d_out;                  // [1,4096,1024] fp32

  // workspace layout (all fully overwritten every call):
  //   Q  bf16 [16][4096][64]   8 MB
  //   K  bf16 [16][4096][64]   8 MB
  //   Vt bf16 [16][64][4096]   8 MB
  //   O  fp32 [4096][1024]    16 MB
  const size_t HEAD_ELEMS = (size_t)NHEAD * T_SEQ * DKHEAD;   // 4,194,304
  unsigned short* Q  = (unsigned short*)d_ws;
  unsigned short* Kb = Q  + HEAD_ELEMS;
  unsigned short* Vt = Kb + HEAD_ELEMS;
  float*          O  = (float*)(Vt + HEAD_ELEMS);

  qkv_gemm_kernel<<<dim3(T_SEQ / 128, 3 * DMODEL / 128), 256, 0, stream>>>(x, Wqkv, Q, Kb, Vt);
  attn_kernel    <<<dim3(T_SEQ / 128, NHEAD),            256, 0, stream>>>(Q, Kb, Vt, O);
  out_gemm_kernel<<<dim3(T_SEQ / 128, DMODEL / 128),     256, 0, stream>>>(O, Wout, out);
}
